// CRF_Layer_20160576487690
// MI455X (gfx1250) — matrix operand order repaired and checked by reading, not compiled, not run on hardware
//
#include <hip/hip_runtime.h>
#include <hip/hip_bf16.h>

// CRF forward-backward + analytic gradients for MI455X (gfx1250).
//   inputs:  data [B,256,128] f32, labels [B,256] int, W [26,128] f32, T [26,26] f32
//   output:  concat(mean_dw [26*128], mean_dT [26*26]) f32  (4004 floats)
//
// Strategy (see analysis): one workgroup per word; WMMA f16 (f32 acc) for the
// two GEMMs; wave0/wave1 run the forward/backward log-space DP concurrently
// (one lane per label, wave32); pairwise marginals normalized by a single logZ.

#define K_LAB 26
#define DFEAT 128
#define MLEN  256
#define KPAD  28      // padded stride for [MLEN][K] f32 LDS arrays

typedef __attribute__((ext_vector_type(16))) _Float16 v16h;
typedef __attribute__((ext_vector_type(8)))  float    v8f;

__device__ __forceinline__ v8f wmma_f16(v16h a, v16h b, v8f c) {
    // v_wmma_f32_16x16x32_f16  (codegen-confirmed builtin, 8 args)
    return __builtin_amdgcn_wmma_f32_16x16x32_f16(
        /*neg_a=*/false, a, /*neg_b=*/false, b,
        /*c_mod=*/(short)0, c, /*reuse_a=*/false, /*reuse_b=*/false);
}

// 16-bit A-matrix 16x32 layout (ISA 7.12.2): lane (0..15)=row M, hi half-wave
// shifts K by 8; VGPRs 0..3 hold K 0..7 / VGPRs 4..7 hold K 16..23 (pairs).
// element e = 2*j + t  ->  kk = (e&8)<<1 | hi<<3 | (e&7)
__device__ __forceinline__ int kkA(int e, int hi) {
    return ((e & 8) << 1) | (hi << 3) | (e & 7);
}
// 16-bit B-matrix 32x16 layout: lane (0..15)=col N; same K order as A (hi half shifts K by 8).
__device__ __forceinline__ int kkB(int e, int hi) {
    return ((e & 8) << 1) | (hi << 3) | (e & 7);
}

// LDS budget (bytes)
#define SH_F32   (3 * MLEN * KPAD + K_LAB * K_LAB + 1)       // sc, la, rb, T, logZ
#define SH_BYTES (SH_F32 * 4 + MLEN * 4 + (32 * DFEAT + 32 * MLEN) * 2)

__global__ void crf_zero_out(float* out, int n) {
    int i = blockIdx.x * blockDim.x + threadIdx.x;
    if (i < n) out[i] = 0.0f;
}

__global__ __launch_bounds__(256) void crf_word_kernel(
    const float* __restrict__ data, const int* __restrict__ labels,
    const float* __restrict__ Wg, const float* __restrict__ Tg,
    float* __restrict__ out, float invB)
{
    extern __shared__ char smem[];
    float*    sc  = (float*)smem;            // [MLEN][KPAD] scores
    float*    la  = sc + MLEN * KPAD;        // [MLEN][KPAD] alpha+scores
    float*    rb  = la + MLEN * KPAD;        // [MLEN][KPAD] beta+scores
    float*    Tl  = rb + MLEN * KPAD;        // [26*26]
    float*    lZ  = Tl + K_LAB * K_LAB;      // [1]
    int*      lab = (int*)(lZ + 1);          // [MLEN]
    _Float16* wh  = (_Float16*)(lab + MLEN); // [32][128] f16 W (rows >=26 zero)
    _Float16* r1h = wh + 32 * DFEAT;         // [32][256] f16 residual^T (rows >=26 zero)

    const int tid  = threadIdx.x;
    const int wave = tid >> 5;
    const int lane = tid & 31;
    const int r    = lane & 15;
    const int hi   = lane >> 4;
    const int word = blockIdx.x;

    const float* xg = data + (size_t)word * MLEN * DFEAT;

    // ---- init: stage T, labels, f16 W ----
    for (int p = tid; p < K_LAB * K_LAB; p += 256) Tl[p] = Tg[p];
    lab[tid] = labels[(size_t)word * MLEN + tid];
    for (int e = tid; e < 32 * DFEAT; e += 256) {
        int row = e >> 7, col = e & 127;
        wh[e] = (row < K_LAB) ? (_Float16)Wg[row * DFEAT + col] : (_Float16)0.0f;
    }
    __syncthreads();

    // ---- phase A: scores = x @ W^T via WMMA ----
    // tiles: 16 M-tiles (rows of x) x 2 N-tiles (labels, padded to 32); K = 128 = 4x32
    for (int s = 0; s < 4; ++s) {
        int tile  = wave + 8 * s;      // 0..31
        int mtile = tile >> 1;
        int ntile = tile & 1;
        v8f acc = {};
        #pragma unroll
        for (int kc = 0; kc < 4; ++kc) {
            v16h a, b;
            const float*    xrow = xg + (size_t)(mtile * 16 + r) * DFEAT + kc * 32;
            const _Float16* wrow = wh + (ntile * 16 + r) * DFEAT + kc * 32;
            #pragma unroll
            for (int e = 0; e < 16; ++e) {
                a[e] = (_Float16)xrow[kkA(e, hi)];
                b[e] = wrow[kkB(e, hi)];
            }
            acc = wmma_f16(a, b, acc);
        }
        #pragma unroll
        for (int rr = 0; rr < 8; ++rr) {   // C layout: M = rr + 8*hi, N = r
            int i = mtile * 16 + hi * 8 + rr;
            int k = ntile * 16 + r;
            if (k < K_LAB) sc[i * KPAD + k] = acc[rr];
        }
    }
    __syncthreads();

    // ---- phase B: log-space DP. wave0: la (forward), wave1: rb (backward) ----
    float Tc[K_LAB];                   // per-lane cached T column/row
    const int y = lane;
    if (wave == 0 && y < K_LAB) {
        #pragma unroll
        for (int yp = 0; yp < K_LAB; ++yp) Tc[yp] = Tl[yp * K_LAB + y]; // T[:,y]
        la[0 * KPAD + y] = sc[0 * KPAD + y];
    }
    if (wave == 1 && y < K_LAB) {
        #pragma unroll
        for (int b2 = 0; b2 < K_LAB; ++b2) Tc[b2] = Tl[y * K_LAB + b2]; // T[y,:]
        rb[(MLEN - 1) * KPAD + y] = sc[(MLEN - 1) * KPAD + y];
    }
    __syncthreads();

    for (int step = 1; step < MLEN; ++step) {
        if (wave == 0 && y < K_LAB) {
            // la[i][y] = sc[i][y] + lse_{y'}( la[i-1][y'] + T[y',y] )
            float v[K_LAB];
            #pragma unroll
            for (int yp = 0; yp < K_LAB; ++yp) v[yp] = la[(step - 1) * KPAD + yp] + Tc[yp];
            float mx = v[0];
            #pragma unroll
            for (int yp = 1; yp < K_LAB; ++yp) mx = fmaxf(mx, v[yp]);
            float sum = 0.0f;
            #pragma unroll
            for (int yp = 0; yp < K_LAB; ++yp) sum += __expf(v[yp] - mx);
            la[step * KPAD + y] = sc[step * KPAD + y] + mx + __logf(sum);
        }
        if (wave == 1 && y < K_LAB) {
            // rb[i][y] = sc[i][y] + lse_b( T[y,b] + rb[i+1][b] )
            int i = MLEN - 1 - step;
            float v[K_LAB];
            #pragma unroll
            for (int b2 = 0; b2 < K_LAB; ++b2) v[b2] = rb[(i + 1) * KPAD + b2] + Tc[b2];
            float mx = v[0];
            #pragma unroll
            for (int b2 = 1; b2 < K_LAB; ++b2) mx = fmaxf(mx, v[b2]);
            float sum = 0.0f;
            #pragma unroll
            for (int b2 = 0; b2 < K_LAB; ++b2) sum += __expf(v[b2] - mx);
            rb[i * KPAD + y] = sc[i * KPAD + y] + mx + __logf(sum);
        }
        __syncthreads();
    }

    // logZ = lse_k la[m-1][k]
    if (tid == 0) {
        float mx = la[(MLEN - 1) * KPAD + 0];
        for (int k = 1; k < K_LAB; ++k) mx = fmaxf(mx, la[(MLEN - 1) * KPAD + k]);
        float sum = 0.0f;
        for (int k = 0; k < K_LAB; ++k) sum += __expf(la[(MLEN - 1) * KPAD + k] - mx);
        lZ[0] = mx + __logf(sum);
    }
    __syncthreads();

    // ---- phase C1: residual r1 = onehot - p1 (f16, transposed), and dT ----
    {
        const float lz = lZ[0];
        const int   i  = tid;              // one position per thread
        const int   yi = lab[i];
        #pragma unroll
        for (int k = 0; k < K_LAB; ++k) {
            // p1 logits = alpha+beta+scores - logZ = la + rb - sc - logZ
            float p = __expf(la[i * KPAD + k] + rb[i * KPAD + k] - sc[i * KPAD + k] - lz);
            r1h[k * MLEN + i] = (_Float16)((k == yi ? 1.0f : 0.0f) - p);
        }
        #pragma unroll
        for (int k = K_LAB; k < 32; ++k) r1h[k * MLEN + i] = (_Float16)0.0f;

        // dT[a,b] = sum_i 1{y_i=a,y_{i+1}=b} - exp(la[i][a] + T[a,b] + rb[i+1][b] - logZ)
        for (int p = tid; p < K_LAB * K_LAB; p += 256) {
            int a = p / K_LAB, b = p % K_LAB;
            float Tab = Tl[p];
            float acc = 0.0f;
            for (int i2 = 0; i2 < MLEN - 1; ++i2) {
                acc -= __expf(la[i2 * KPAD + a] + rb[(i2 + 1) * KPAD + b] + Tab - lz);
                acc += (lab[i2] == a && lab[i2 + 1] == b) ? 1.0f : 0.0f;
            }
            atomicAdd(&out[K_LAB * DFEAT + p], acc * invB);
        }
    }
    __syncthreads();

    // ---- phase C2: dw = r1^T @ x via WMMA (M=32 pad, N=128, K=256) ----
    for (int s = 0; s < 2; ++s) {
        int tile  = wave * 2 + s;      // 0..15
        int mtile = tile >> 3;         // label block
        int ntile = tile & 7;          // feature block
        v8f acc = {};
        #pragma unroll
        for (int kc = 0; kc < 8; ++kc) {
            v16h a, b;
            const _Float16* arow = r1h + (mtile * 16 + r) * MLEN + kc * 32;
            const float*    bcol = xg + (size_t)(kc * 32) * DFEAT + ntile * 16 + r;
            #pragma unroll
            for (int e = 0; e < 16; ++e) {
                a[e] = arow[kkA(e, hi)];
                b[e] = (_Float16)bcol[(size_t)kkB(e, hi) * DFEAT];
            }
            acc = wmma_f16(a, b, acc);
        }
        #pragma unroll
        for (int rr = 0; rr < 8; ++rr) {
            int k = mtile * 16 + hi * 8 + rr;
            int d = ntile * 16 + r;
            if (k < K_LAB) atomicAdd(&out[k * DFEAT + d], acc[rr] * invB);
        }
    }
}

extern "C" void kernel_launch(void* const* d_in, const int* in_sizes, int n_in,
                              void* d_out, int out_size, void* d_ws, size_t ws_size,
                              hipStream_t stream) {
    const float* data   = (const float*)d_in[0];
    const int*   labels = (const int*)d_in[1];
    const float* Wg     = (const float*)d_in[2];
    const float* Tg     = (const float*)d_in[3];
    float*       out    = (float*)d_out;

    const int B    = in_sizes[1] / MLEN;     // number of words
    const float iB = 1.0f / (float)B;

    // dynamic LDS > 64KB: raise the limit (non-stream API, capture-safe)
    hipFuncSetAttribute((const void*)crf_word_kernel,
                        hipFuncAttributeMaxDynamicSharedMemorySize, SH_BYTES);

    crf_zero_out<<<(out_size + 255) / 256, 256, 0, stream>>>(out, out_size);
    crf_word_kernel<<<B, 256, SH_BYTES, stream>>>(data, labels, Wg, Tg, out, iB);
}
